// GotenNet_85323820302759
// MI455X (gfx1250) — matrix operand order repaired and checked by reading, not compiled, not run on hardware
//
#include <hip/hip_runtime.h>

// Indices are cast per harness convention (integer -> const int*).

#define N_NODES 100000
#define E1N     200000
#define E2N     400000
#define DD      128
#define HH      8
#define SM_BLOCKS 512

typedef __attribute__((ext_vector_type(16))) __bf16 v16bf;
typedef __attribute__((ext_vector_type(8)))  float  v8f;

#define WMMA_BF16(acc, a, b) \
  (acc) = __builtin_amdgcn_wmma_f32_16x16x32_bf16(false,(a),false,(b),(short)0,(acc),false,false)

// ---- bf16 weight region: element offsets inside ws[0 .. 1MB) ----
constexpr size_t WQT  = 0;
constexpr size_t WKT  = 16384;
constexpr size_t WVT  = 32768;
constexpr size_t WOT  = 49152;
constexpr size_t WGT  = 65536;          // 16x128 (heads padded 8->16 with zeros)
constexpr size_t WE1T = 67584;          // 128x384
constexpr size_t WE2T = 116736;         // 128x128
constexpr size_t WH1T = 133120;         // 128x384
constexpr size_t WH2T = 182272;         // 128x128
constexpr size_t WF1T = 198656;         // 256x128
constexpr size_t WF2T = 231424;         // 128x256

// ---- workspace byte offsets ----
constexpr size_t OFF_LOGITS = (size_t)1 << 20;                    // E2*8 f32 (12.8 MB)
constexpr size_t OFF_VBUF   = OFF_LOGITS + ((size_t)14 << 20);    // E2*128 bf16 (102.4 MB)
constexpr size_t OFF_HUPD   = OFF_VBUF   + ((size_t)98 << 20);    // N*128 f32 (51.2 MB)
constexpr size_t OFF_HNEW   = OFF_HUPD   + ((size_t)49 << 20);    // N*128 f32 (51.2 MB)
constexpr size_t OFF_PART   = OFF_HNEW   + ((size_t)49 << 20);    // softmax partials

// ---------------- fragment helpers (layouts per CDNA5 ISA 7.12.2) ----------------
// A (16x32 bf16): lane l row M=l%16; element j -> K = k0 + 16*(j/8) + 8*(l/16) + (j%8)
__device__ __forceinline__ v16bf a_frag_f32(const float* __restrict__ row, int k0, int hi) {
  const float* p0 = row + k0 + hi * 8;
  const float* p1 = p0 + 16;
  v16bf a;
#pragma unroll
  for (int j = 0; j < 8; ++j) { a[j] = (__bf16)p0[j]; a[j + 8] = (__bf16)p1[j]; }
  return a;
}
__device__ __forceinline__ v16bf a_frag_bf16(const __bf16* __restrict__ row, int k0, int hi) {
  const __bf16* p0 = row + k0 + hi * 8;
  const __bf16* p1 = p0 + 16;
  v16bf a;
#pragma unroll
  for (int j = 0; j < 8; ++j) { a[j] = p0[j]; a[j + 8] = p1[j]; }
  return a;
}
// B (32x16 bf16): lane l col N; element j -> K = k0 + 16*(j/8) + 8*(l/16) + (j%8).
// Wt is the transposed weight: Wt[n*ldk + k] == W[k][n] -> contiguous per-lane run.
__device__ __forceinline__ v16bf b_frag(const __bf16* __restrict__ Wt, int ldk,
                                        int n, int k0, int hi) {
  const __bf16* p = Wt + (size_t)n * ldk + k0 + 8 * hi;
  v16bf b;
#pragma unroll
  for (int j = 0; j < 8; ++j) { b[j] = p[j]; b[j + 8] = p[j + 16]; }
  return b;
}
// sum across the 16 lanes of a half-wave (ds_swizzle xor butterflies)
__device__ __forceinline__ float red16(float x) {
  x += __int_as_float(__builtin_amdgcn_ds_swizzle(__float_as_int(x), 0x041F)); // xor 1
  x += __int_as_float(__builtin_amdgcn_ds_swizzle(__float_as_int(x), 0x081F)); // xor 2
  x += __int_as_float(__builtin_amdgcn_ds_swizzle(__float_as_int(x), 0x101F)); // xor 4
  x += __int_as_float(__builtin_amdgcn_ds_swizzle(__float_as_int(x), 0x201F)); // xor 8
  return x;
}
__device__ __forceinline__ float silu(float x) { return x / (1.f + __expf(-x)); }

// ---------------- weight prep ----------------
__global__ void k_prep_w(const float* __restrict__ W, __bf16* __restrict__ Wt, int K, int N) {
  int i = blockIdx.x * blockDim.x + threadIdx.x;
  if (i < K * N) { int k = i / N, n = i % N; Wt[(size_t)n * K + k] = (__bf16)W[i]; }
}
__global__ void k_prep_wg(const float* __restrict__ Wg, __bf16* __restrict__ wgt) {
  int i = blockIdx.x * blockDim.x + threadIdx.x;         // 16*128
  if (i < 16 * 128) {
    int n = i / 128, k = i % 128;
    wgt[i] = (__bf16)((n < HH) ? Wg[k * HH + n] : 0.f);
  }
}
__global__ void k_zero(float* __restrict__ p, long n) {
  long i = (long)blockIdx.x * blockDim.x + threadIdx.x;
  if (i < n) p[i] = 0.f;
}

// ---------------- stage 1: q,k,v projections + attention logits ----------------
__global__ void __launch_bounds__(128) k_qkv_logits(
    const float* __restrict__ h, const float* __restrict__ t_e2,
    const int* __restrict__ ei2,
    const float* __restrict__ bq, const float* __restrict__ bk,
    const float* __restrict__ bv, const float* __restrict__ bg,
    const __bf16* __restrict__ wb,
    float* __restrict__ logits, __bf16* __restrict__ vbuf)
{
  const __bf16* wq = wb + WQT;
  const __bf16* wk = wb + WKT;
  const __bf16* wv = wb + WVT;
  const __bf16* wg = wb + WGT;
  const int lane = threadIdx.x & 31, wave = threadIdx.x >> 5;
  const int l16 = lane & 15, hi = lane >> 4;
  const int ntiles = E2N / 16;
  int tile = blockIdx.x * 4 + wave;
  if (tile >= ntiles) return;                 // wave-uniform, no barriers here
  const size_t eb = (size_t)tile * 16;

  size_t eA = eb + l16; if (eA > (size_t)E2N - 1) eA = E2N - 1;
  const int srcA = ei2[eA];
  const int dstA = ei2[(size_t)E2N + eA];
  const float* rq = h + (size_t)dstA * DD;
  const float* rk = h + (size_t)srcA * DD;
  const float* rt = t_e2 + eA * DD;

  v8f qa[8], ka[8], va[8], ga;
#pragma unroll
  for (int t = 0; t < 8; ++t)
#pragma unroll
    for (int j = 0; j < 8; ++j) { qa[t][j] = 0.f; ka[t][j] = 0.f; va[t][j] = 0.f; }
#pragma unroll
  for (int j = 0; j < 8; ++j) ga[j] = 0.f;

#pragma unroll
  for (int k0 = 0; k0 < DD; k0 += 32) {
    v16bf aq = a_frag_f32(rq, k0, hi);
    v16bf ak = a_frag_f32(rk, k0, hi);
    v16bf at = a_frag_f32(rt, k0, hi);
#pragma unroll
    for (int t = 0; t < 8; ++t) {
      WMMA_BF16(qa[t], aq, b_frag(wq, DD, t * 16 + l16, k0, hi));
      WMMA_BF16(ka[t], ak, b_frag(wk, DD, t * 16 + l16, k0, hi));
      WMMA_BF16(va[t], ak, b_frag(wv, DD, t * 16 + l16, k0, hi));
    }
    WMMA_BF16(ga, at, b_frag(wg, DD, l16, k0, hi));
  }

  // store v (+bias) as bf16
#pragma unroll
  for (int t = 0; t < 8; ++t) {
    const float bvv = bv[t * 16 + l16];
#pragma unroll
    for (int r = 0; r < 8; ++r) {
      const size_t e = eb + r + 8 * hi;
      vbuf[e * DD + t * 16 + l16] = (__bf16)(va[t][r] + bvv);
    }
  }

  // logits[e][hd] = (q.k)/4 + t@Wg + bg ; head hd == n-tile hd (Dh==16)
  float sreg[8];
#pragma unroll
  for (int r = 0; r < 8; ++r) sreg[r] = 0.f;
#pragma unroll
  for (int hd = 0; hd < 8; ++hd) {
    const float bqv = bq[hd * 16 + l16];
    const float bkv = bk[hd * 16 + l16];
#pragma unroll
    for (int r = 0; r < 8; ++r) {
      float x = (qa[hd][r] + bqv) * (ka[hd][r] + bkv);
      x = red16(x);
      if (l16 == hd) sreg[r] = x;
    }
  }
  if (l16 < HH) {
    const float bgv = bg[l16];
#pragma unroll
    for (int r = 0; r < 8; ++r) {
      const size_t e = eb + r + 8 * hi;
      logits[e * HH + l16] = sreg[r] * 0.25f + ga[r] + bgv;
    }
  }
}

// ---------------- stage 2: global softmax over edge axis ----------------
__global__ void k_sm_partial(const float* __restrict__ logits, float* __restrict__ partials) {
  __shared__ float sm[256], ss[256];
  const int tid = threadIdx.x, hd = tid & 7, slot = tid >> 3;
  const long chunk = (E2N + SM_BLOCKS - 1) / SM_BLOCKS;
  const long e0 = (long)blockIdx.x * chunk;
  long eend = e0 + chunk; if (eend > E2N) eend = E2N;
  float m = -1e30f, s = 0.f;
  for (long e = e0 + slot; e < eend; e += 32) {
    float x = logits[e * HH + hd];
    float m2 = fmaxf(m, x);
    s = s * __expf(m - m2) + __expf(x - m2);
    m = m2;
  }
  sm[tid] = m; ss[tid] = s;
  __syncthreads();
  for (int off = 128; off >= 8; off >>= 1) {
    if (tid < off) {
      float m1 = sm[tid], s1 = ss[tid], m2 = sm[tid + off], s2 = ss[tid + off];
      float mm = fmaxf(m1, m2);
      sm[tid] = mm;
      ss[tid] = s1 * __expf(m1 - mm) + s2 * __expf(m2 - mm);
    }
    __syncthreads();
  }
  if (tid < 8) {
    partials[((size_t)blockIdx.x * 8 + tid) * 2]     = sm[tid];
    partials[((size_t)blockIdx.x * 8 + tid) * 2 + 1] = ss[tid];
  }
}
__global__ void k_sm_final(const float* __restrict__ partials, float* __restrict__ stats) {
  int hd = threadIdx.x;
  if (hd >= 8) return;
  float m = -1e30f, s = 0.f;
  for (int b = 0; b < SM_BLOCKS; ++b) {
    float m2 = partials[((size_t)b * 8 + hd) * 2];
    float s2 = partials[((size_t)b * 8 + hd) * 2 + 1];
    float mm = fmaxf(m, m2);
    s = s * __expf(m - mm) + s2 * __expf(m2 - mm);
    m = mm;
  }
  stats[hd] = m;
  stats[8 + hd] = (s > 0.f) ? 1.f / s : 0.f;
}

// ---------------- stage 3: msg = alpha*v scatter-added to h_update ----------------
__global__ void k_scatter(const float* __restrict__ logits, const __bf16* __restrict__ vbuf,
                          const int* __restrict__ ei2, const float* __restrict__ stats,
                          float* __restrict__ hupd) {
  const size_t e = (size_t)blockIdx.x * 2 + (threadIdx.x >> 7);
  const int d = threadIdx.x & 127;
  if (e >= E2N) return;
  const int hd = d >> 4;
  const float a = __expf(logits[e * HH + hd] - stats[hd]) * stats[8 + hd];
  const float val = a * (float)vbuf[e * DD + d];
  const int dst = ei2[(size_t)E2N + e];
  atomicAdd(&hupd[(size_t)dst * DD + d], val);
}

// ---------------- stage 4: h_new = h + h_update @ Wo + bo ----------------
__global__ void __launch_bounds__(128) k_update_h(
    const float* __restrict__ h, const float* __restrict__ hupd,
    const float* __restrict__ bo, const __bf16* __restrict__ wb,
    float* __restrict__ hnew)
{
  const __bf16* wo = wb + WOT;
  const int lane = threadIdx.x & 31, wave = threadIdx.x >> 5;
  const int l16 = lane & 15, hi = lane >> 4;
  const int ntiles = N_NODES / 16;
  int tile = blockIdx.x * 4 + wave;
  if (tile >= ntiles) return;
  const size_t rb = (size_t)tile * 16;
  size_t rA = rb + l16; if (rA > (size_t)N_NODES - 1) rA = N_NODES - 1;
  const float* arow = hupd + rA * DD;

  v8f acc[8];
#pragma unroll
  for (int t = 0; t < 8; ++t)
#pragma unroll
    for (int j = 0; j < 8; ++j) acc[t][j] = 0.f;
#pragma unroll
  for (int k0 = 0; k0 < DD; k0 += 32) {
    v16bf af = a_frag_f32(arow, k0, hi);
#pragma unroll
    for (int t = 0; t < 8; ++t)
      WMMA_BF16(acc[t], af, b_frag(wo, DD, t * 16 + l16, k0, hi));
  }
#pragma unroll
  for (int t = 0; t < 8; ++t) {
    const int c = t * 16 + l16;
    const float bov = bo[c];
#pragma unroll
    for (int r = 0; r < 8; ++r) {
      const size_t rr = rb + r + 8 * hi;
      hnew[rr * DD + c] = h[rr * DD + c] + acc[t][r] + bov;
    }
  }
}

// ---------------- stage 5: t_e2 += MLP(concat[h_s, h_d, t]) ----------------
__global__ void __launch_bounds__(128) k_edge_mlp(
    const float* __restrict__ hnew, const float* __restrict__ t_e2,
    const int* __restrict__ ei2,
    const float* __restrict__ be1, const float* __restrict__ be2,
    const __bf16* __restrict__ wb, float* __restrict__ tout)
{
  __shared__ __bf16 hid[4][16 * DD];
  const __bf16* w1 = wb + WE1T;
  const __bf16* w2 = wb + WE2T;
  const int lane = threadIdx.x & 31, wave = threadIdx.x >> 5;
  const int l16 = lane & 15, hi = lane >> 4;
  const int ntiles = E2N / 16;
  int tile = blockIdx.x * 4 + wave;
  const bool wvalid = tile < ntiles;
  if (!wvalid) tile = ntiles - 1;
  const size_t eb = (size_t)tile * 16;

  size_t eA = eb + l16; if (eA > (size_t)E2N - 1) eA = E2N - 1;
  const float* rs = hnew + (size_t)ei2[eA] * DD;
  const float* rd = hnew + (size_t)ei2[(size_t)E2N + eA] * DD;
  const float* rt = t_e2 + eA * DD;

  v8f acc[8];
#pragma unroll
  for (int t = 0; t < 8; ++t)
#pragma unroll
    for (int j = 0; j < 8; ++j) acc[t][j] = 0.f;
#pragma unroll
  for (int k0 = 0; k0 < 3 * DD; k0 += 32) {
    const float* rp; int kk;
    if (k0 < DD)          { rp = rs; kk = k0; }
    else if (k0 < 2 * DD) { rp = rd; kk = k0 - DD; }
    else                  { rp = rt; kk = k0 - 2 * DD; }
    v16bf af = a_frag_f32(rp, kk, hi);
#pragma unroll
    for (int t = 0; t < 8; ++t)
      WMMA_BF16(acc[t], af, b_frag(w1, 3 * DD, t * 16 + l16, k0, hi));
  }
  __bf16* hrow = hid[wave];
#pragma unroll
  for (int t = 0; t < 8; ++t) {
    const int c = t * 16 + l16;
    const float b1 = be1[c];
#pragma unroll
    for (int r = 0; r < 8; ++r)
      hrow[(r + 8 * hi) * DD + c] = (__bf16)silu(acc[t][r] + b1);
  }
  __syncthreads();

  v8f acc2[8];
#pragma unroll
  for (int t = 0; t < 8; ++t)
#pragma unroll
    for (int j = 0; j < 8; ++j) acc2[t][j] = 0.f;
  const __bf16* arow2 = hrow + l16 * DD;
#pragma unroll
  for (int k0 = 0; k0 < DD; k0 += 32) {
    v16bf af = a_frag_bf16(arow2, k0, hi);
#pragma unroll
    for (int t = 0; t < 8; ++t)
      WMMA_BF16(acc2[t], af, b_frag(w2, DD, t * 16 + l16, k0, hi));
  }
#pragma unroll
  for (int t = 0; t < 8; ++t) {
    const int c = t * 16 + l16;
    const float b2 = be2[c];
#pragma unroll
    for (int r = 0; r < 8; ++r) {
      const size_t e = eb + r + 8 * hi;
      if (wvalid) tout[e * DD + c] = t_e2[e * DD + c] + acc2[t][r] + b2;
    }
  }
}

// ---------------- stage 6: EdgeHTR refresh of E1 sub-edges (in place on tout) -------
__global__ void __launch_bounds__(128) k_edgehtr(
    const float* __restrict__ hnew, const int* __restrict__ ei1,
    const int* __restrict__ e1map,
    const float* __restrict__ bh1, const float* __restrict__ bh2,
    const __bf16* __restrict__ wb, float* __restrict__ tout)
{
  __shared__ __bf16 hid[4][16 * DD];
  const __bf16* w1 = wb + WH1T;
  const __bf16* w2 = wb + WH2T;
  const int lane = threadIdx.x & 31, wave = threadIdx.x >> 5;
  const int l16 = lane & 15, hi = lane >> 4;
  const int ntiles = E1N / 16;
  int tile = blockIdx.x * 4 + wave;
  const bool wvalid = tile < ntiles;
  if (!wvalid) tile = ntiles - 1;
  const size_t eb = (size_t)tile * 16;

  size_t eA = eb + l16; if (eA > (size_t)E1N - 1) eA = E1N - 1;
  const float* rs = hnew + (size_t)ei1[eA] * DD;
  const float* rd = hnew + (size_t)ei1[(size_t)E1N + eA] * DD;
  const float* rt = tout + (size_t)e1map[eA] * DD;

  v8f acc[8];
#pragma unroll
  for (int t = 0; t < 8; ++t)
#pragma unroll
    for (int j = 0; j < 8; ++j) acc[t][j] = 0.f;
#pragma unroll
  for (int k0 = 0; k0 < 3 * DD; k0 += 32) {
    const float* rp; int kk;
    if (k0 < DD)          { rp = rs; kk = k0; }
    else if (k0 < 2 * DD) { rp = rd; kk = k0 - DD; }
    else                  { rp = rt; kk = k0 - 2 * DD; }
    v16bf af = a_frag_f32(rp, kk, hi);
#pragma unroll
    for (int t = 0; t < 8; ++t)
      WMMA_BF16(acc[t], af, b_frag(w1, 3 * DD, t * 16 + l16, k0, hi));
  }
  __bf16* hrow = hid[wave];
#pragma unroll
  for (int t = 0; t < 8; ++t) {
    const int c = t * 16 + l16;
    const float b1 = bh1[c];
#pragma unroll
    for (int r = 0; r < 8; ++r)
      hrow[(r + 8 * hi) * DD + c] = (__bf16)silu(acc[t][r] + b1);
  }
  __syncthreads();

  v8f acc2[8];
#pragma unroll
  for (int t = 0; t < 8; ++t)
#pragma unroll
    for (int j = 0; j < 8; ++j) acc2[t][j] = 0.f;
  const __bf16* arow2 = hrow + l16 * DD;
#pragma unroll
  for (int k0 = 0; k0 < DD; k0 += 32) {
    v16bf af = a_frag_bf16(arow2, k0, hi);
#pragma unroll
    for (int t = 0; t < 8; ++t)
      WMMA_BF16(acc2[t], af, b_frag(w2, DD, t * 16 + l16, k0, hi));
  }
#pragma unroll
  for (int t = 0; t < 8; ++t) {
    const int c = t * 16 + l16;
    const float b2 = bh2[c];
#pragma unroll
    for (int r = 0; r < 8; ++r) {
      const size_t e = eb + r + 8 * hi;
      if (wvalid && e < E1N) {
        const size_t e2 = (size_t)e1map[e];
        const float old = tout[e2 * DD + c];     // sub_t (written by k_edge_mlp)
        tout[e2 * DD + c] = old + acc2[t][r] + b2;
      }
    }
  }
}

// ---------------- stage 7: EQFF h_out = h_new + MLP(h_new) ----------------
__global__ void __launch_bounds__(128) k_eqff(
    const float* __restrict__ hnew,
    const float* __restrict__ bf1, const float* __restrict__ bf2,
    const __bf16* __restrict__ wb, float* __restrict__ hout)
{
  __shared__ __bf16 hid[4][16 * 256];
  const __bf16* w1 = wb + WF1T;
  const __bf16* w2 = wb + WF2T;
  const int lane = threadIdx.x & 31, wave = threadIdx.x >> 5;
  const int l16 = lane & 15, hi = lane >> 4;
  const int ntiles = N_NODES / 16;
  int tile = blockIdx.x * 4 + wave;
  const bool wvalid = tile < ntiles;
  if (!wvalid) tile = ntiles - 1;
  const size_t rb = (size_t)tile * 16;
  size_t rA = rb + l16; if (rA > (size_t)N_NODES - 1) rA = N_NODES - 1;
  const float* arow = hnew + rA * DD;

  v8f acc[16];
#pragma unroll
  for (int t = 0; t < 16; ++t)
#pragma unroll
    for (int j = 0; j < 8; ++j) acc[t][j] = 0.f;
#pragma unroll
  for (int k0 = 0; k0 < DD; k0 += 32) {
    v16bf af = a_frag_f32(arow, k0, hi);
#pragma unroll
    for (int t = 0; t < 16; ++t)
      WMMA_BF16(acc[t], af, b_frag(w1, DD, t * 16 + l16, k0, hi));
  }
  __bf16* hrow = hid[wave];
#pragma unroll
  for (int t = 0; t < 16; ++t) {
    const int c = t * 16 + l16;
    const float b1 = bf1[c];
#pragma unroll
    for (int r = 0; r < 8; ++r)
      hrow[(r + 8 * hi) * 256 + c] = (__bf16)silu(acc[t][r] + b1);
  }
  __syncthreads();

  v8f acc2[8];
#pragma unroll
  for (int t = 0; t < 8; ++t)
#pragma unroll
    for (int j = 0; j < 8; ++j) acc2[t][j] = 0.f;
  const __bf16* arow2 = hrow + l16 * 256;
#pragma unroll
  for (int k0 = 0; k0 < 256; k0 += 32) {
    v16bf af = a_frag_bf16(arow2, k0, hi);
#pragma unroll
    for (int t = 0; t < 8; ++t)
      WMMA_BF16(acc2[t], af, b_frag(w2, 256, t * 16 + l16, k0, hi));
  }
#pragma unroll
  for (int t = 0; t < 8; ++t) {
    const int c = t * 16 + l16;
    const float b2 = bf2[c];
#pragma unroll
    for (int r = 0; r < 8; ++r) {
      const size_t rr = rb + r + 8 * hi;
      if (wvalid) hout[rr * DD + c] = hnew[rr * DD + c] + acc2[t][r] + b2;
    }
  }
}

// ---------------- host launcher ----------------
extern "C" void kernel_launch(void* const* d_in, const int* in_sizes, int n_in,
                              void* d_out, int out_size, void* d_ws, size_t ws_size,
                              hipStream_t stream) {
  (void)in_sizes; (void)n_in; (void)out_size; (void)ws_size;
  const float* h    = (const float*)d_in[0];
  const float* t_e2 = (const float*)d_in[1];
  const int*   ei1  = (const int*)d_in[2];
  const int*   ei2  = (const int*)d_in[3];
  const int*   e1m  = (const int*)d_in[4];
  const float* Wq = (const float*)d_in[5];  const float* bq = (const float*)d_in[6];
  const float* Wk = (const float*)d_in[7];  const float* bk = (const float*)d_in[8];
  const float* Wv = (const float*)d_in[9];  const float* bv = (const float*)d_in[10];
  const float* Wg = (const float*)d_in[11]; const float* bg = (const float*)d_in[12];
  const float* Wo = (const float*)d_in[13]; const float* bo = (const float*)d_in[14];
  const float* We1 = (const float*)d_in[15]; const float* be1 = (const float*)d_in[16];
  const float* We2 = (const float*)d_in[17]; const float* be2 = (const float*)d_in[18];
  const float* Wh1 = (const float*)d_in[19]; const float* bh1 = (const float*)d_in[20];
  const float* Wh2 = (const float*)d_in[21]; const float* bh2 = (const float*)d_in[22];
  const float* Wf1 = (const float*)d_in[23]; const float* bf1 = (const float*)d_in[24];
  const float* Wf2 = (const float*)d_in[25]; const float* bf2 = (const float*)d_in[26];

  float* hout = (float*)d_out;
  float* tout = hout + (size_t)N_NODES * DD;

  char*   ws     = (char*)d_ws;
  __bf16* wb     = (__bf16*)ws;
  float*  logits = (float*)(ws + OFF_LOGITS);
  __bf16* vbuf   = (__bf16*)(ws + OFF_VBUF);
  float*  hupd   = (float*)(ws + OFF_HUPD);
  float*  hnew   = (float*)(ws + OFF_HNEW);
  float*  parts  = (float*)(ws + OFF_PART);
  float*  stats  = parts + (size_t)SM_BLOCKS * 16;

  auto prep = [&](const float* W, size_t off, int K, int N) {
    int n = K * N;
    k_prep_w<<<(n + 255) / 256, 256, 0, stream>>>(W, wb + off, K, N);
  };
  prep(Wq, WQT, 128, 128);  prep(Wk, WKT, 128, 128);
  prep(Wv, WVT, 128, 128);  prep(Wo, WOT, 128, 128);
  k_prep_wg<<<(16 * 128 + 255) / 256, 256, 0, stream>>>(Wg, wb + WGT);
  prep(We1, WE1T, 384, 128); prep(We2, WE2T, 128, 128);
  prep(Wh1, WH1T, 384, 128); prep(Wh2, WH2T, 128, 128);
  prep(Wf1, WF1T, 128, 256); prep(Wf2, WF2T, 256, 128);

  const int t2 = E2N / 16, t1 = E1N / 16, tn = N_NODES / 16;

  k_qkv_logits<<<(t2 + 3) / 4, 128, 0, stream>>>(h, t_e2, ei2, bq, bk, bv, bg,
                                                 wb, logits, vbuf);
  k_sm_partial<<<SM_BLOCKS, 256, 0, stream>>>(logits, parts);
  k_sm_final<<<1, 32, 0, stream>>>(parts, stats);
  k_zero<<<((long)N_NODES * DD + 255) / 256, 256, 0, stream>>>(hupd, (long)N_NODES * DD);
  k_scatter<<<E2N / 2, 256, 0, stream>>>(logits, vbuf, ei2, stats, hupd);
  k_update_h<<<(tn + 3) / 4, 128, 0, stream>>>(h, hupd, bo, wb, hnew);
  k_edge_mlp<<<(t2 + 3) / 4, 128, 0, stream>>>(hnew, t_e2, ei2, be1, be2, wb, tout);
  k_edgehtr<<<(t1 + 3) / 4, 128, 0, stream>>>(hnew, ei1, e1m, bh1, bh2, wb, tout);
  k_eqff<<<(tn + 3) / 4, 128, 0, stream>>>(hnew, bf1, bf2, wb, hout);
}
